// RetentionEncoderLayer_69569880261096
// MI455X (gfx1250) — matrix operand order repaired and checked by reading, not compiled, not run on hardware
//
#include <hip/hip_runtime.h>
#include <hip/hip_bf16.h>
#include <math.h>

typedef _Float16 f16;
typedef __attribute__((ext_vector_type(16))) _Float16 v16h;
typedef __attribute__((ext_vector_type(8)))  _Float16 v8h;
typedef __attribute__((ext_vector_type(8)))  float    v8f;

#define G_ 32
#define S_ 512
#define D_ 128
#define H_ 8
#define HD_ 16
#define FFN_ 256
#define BM_ (G_ * S_)

#define LOG2E_ 1.4426950408889634f

// ---------------------------------------------------------------- helpers

__device__ __forceinline__ float fast_exp2(float x) {
  return __builtin_amdgcn_exp2f(x);   // v_exp_f32
}

__device__ __forceinline__ v8f zero_v8f() {
  v8f c;
#pragma unroll
  for (int i = 0; i < 8; ++i) c[i] = 0.0f;
  return c;
}

__device__ __forceinline__ v8f wmma_f16(v16h a, v16h b, v8f c) {
  // v_wmma_f32_16x16x32_f16  (neg_a, A, neg_b, B, c_mod, C, reuse_a, reuse_b)
  return __builtin_amdgcn_wmma_f32_16x16x32_f16(false, a, false, b, (short)0, c,
                                                false, false);
}

// A fragment (16x32, f16, K full) from row-major [., ld] at (row0, k0).
// lanes 0-15: M=lane, K = k0+{0..7, 16..23}; lanes 16-31: K = k0+{8..15, 24..31}
__device__ __forceinline__ v16h load_a_frag(const f16* base, int ld, int row0,
                                            int k0) {
  int lane = threadIdx.x & 31;
  int row = row0 + (lane & 15);
  int kb = k0 + ((lane & 16) ? 8 : 0);
  const f16* p = base + (size_t)row * ld + kb;
  v8h lo = *reinterpret_cast<const v8h*>(p);
  v8h hi = *reinterpret_cast<const v8h*>(p + 16);
  v16h a;
#pragma unroll
  for (int i = 0; i < 8; ++i) { a[i] = lo[i]; a[i + 8] = hi[i]; }
  return a;
}

// A fragment with real K=16 (HD), upper K half zero-padded.
__device__ __forceinline__ v16h load_a_frag_k16(const f16* base, int ld,
                                                int row0) {
  int lane = threadIdx.x & 31;
  int row = row0 + (lane & 15);
  int kb = (lane & 16) ? 8 : 0;
  v8h lo = *reinterpret_cast<const v8h*>(base + (size_t)row * ld + kb);
  v16h a;
#pragma unroll
  for (int i = 0; i < 8; ++i) { a[i] = lo[i]; a[i + 8] = (f16)0; }
  return a;
}

// B fragment (32x16) from B^T stored row-major [n][k] with leading dim ld.
// lane n=lane&15; lanes 0-15: K = k0+{0..7, 16..23}; lanes 16-31: K = k0+{8..15, 24..31}
__device__ __forceinline__ v16h load_bt_frag(const f16* bt, int ld, int n0,
                                             int k0) {
  int lane = threadIdx.x & 31;
  int n = n0 + (lane & 15);
  int kb = k0 + ((lane & 16) ? 8 : 0);
  const f16* p = bt + (size_t)n * ld + kb;
  v8h lo = *reinterpret_cast<const v8h*>(p);
  v8h hi = *reinterpret_cast<const v8h*>(p + 16);
  v16h b;
#pragma unroll
  for (int i = 0; i < 8; ++i) { b[i] = lo[i]; b[i + 8] = hi[i]; }
  return b;
}

// B fragment for Q@K^T: B[d][t] = k[t][d]; real K(=d)=16, upper half zeros.
// krows is k stored [t][HD]. lanes 0-15: d=0..7; lanes 16-31: d=8..15; elems 8-15 zero
__device__ __forceinline__ v16h load_b_frag_k16(const f16* krows, int t0) {
  int lane = threadIdx.x & 31;
  v16h b;
  {
    const f16* p = krows + (size_t)(t0 + (lane & 15)) * HD_ + ((lane & 16) ? 8 : 0);
    v8h lo = *reinterpret_cast<const v8h*>(p);
#pragma unroll
    for (int i = 0; i < 8; ++i) { b[i] = lo[i]; b[i + 8] = (f16)0; }
  }
  return b;
}

__device__ __forceinline__ float redsum16(float x) {
  x += __shfl_xor(x, 1, 16);
  x += __shfl_xor(x, 2, 16);
  x += __shfl_xor(x, 4, 16);
  x += __shfl_xor(x, 8, 16);
  return x;
}
__device__ __forceinline__ float redsum32(float x) {
  x += __shfl_xor(x, 1, 32);
  x += __shfl_xor(x, 2, 32);
  x += __shfl_xor(x, 4, 32);
  x += __shfl_xor(x, 8, 32);
  x += __shfl_xor(x, 16, 32);
  return x;
}

// ---------------------------------------------------------------- kernels

// transpose + f32->f16:  dst[c][r] = src[r][c]   (src is R x C row-major)
__global__ void k_wt(const float* __restrict__ src, f16* __restrict__ dst,
                     int R, int C) {
  int i = blockIdx.x * 256 + threadIdx.x;
  if (i < R * C) {
    int r = i / C, c = i % C;
    dst[(size_t)c * R + r] = (f16)src[i];
  }
}

// RMSNorm(x, ln1_w) -> xn (f16);  qin = xn + pe (f16). one wave per row.
__global__ void k_prenorm(const float* __restrict__ x,
                          const float* __restrict__ pe,
                          const float* __restrict__ w, f16* __restrict__ xn,
                          f16* __restrict__ qin) {
  int row = blockIdx.x * 8 + (threadIdx.x >> 5);
  int lane = threadIdx.x & 31;
  const float4 xv = reinterpret_cast<const float4*>(x + (size_t)row * D_)[lane];
  float ss = xv.x * xv.x + xv.y * xv.y + xv.z * xv.z + xv.w * xv.w;
  ss = redsum32(ss);
  float r = rsqrtf(ss * (1.0f / D_) + 1e-6f);
  const float4 wv = reinterpret_cast<const float4*>(w)[lane];
  const float4 pv = reinterpret_cast<const float4*>(pe + (size_t)row * D_)[lane];
  float n0 = xv.x * r * wv.x, n1 = xv.y * r * wv.y;
  float n2 = xv.z * r * wv.z, n3 = xv.w * r * wv.w;
  f16* xo = xn + (size_t)row * D_ + lane * 4;
  f16* qo = qin + (size_t)row * D_ + lane * 4;
  xo[0] = (f16)n0; xo[1] = (f16)n1; xo[2] = (f16)n2; xo[3] = (f16)n3;
  qo[0] = (f16)(n0 + pv.x); qo[1] = (f16)(n1 + pv.y);
  qo[2] = (f16)(n2 + pv.z); qo[3] = (f16)(n3 + pv.w);
}

// Fused Q/K/V/G projections. block = 8 waves = 8 heads, one 16-row M tile.
__global__ void k_qkvg(const f16* __restrict__ qin, const f16* __restrict__ xn,
                       const f16* __restrict__ WtQ, const f16* __restrict__ WtK,
                       const f16* __restrict__ WtV, const f16* __restrict__ WtG,
                       f16* __restrict__ qb, f16* __restrict__ kb,
                       f16* __restrict__ vT, f16* __restrict__ gb) {
  int mt = blockIdx.x;
  int h = threadIdx.x >> 5;
  int lane = threadIdx.x & 31;
  int row0 = mt * 16;
  v8f cq = zero_v8f(), ck = zero_v8f(), cv = zero_v8f(), cg = zero_v8f();
#pragma unroll
  for (int kk = 0; kk < D_; kk += 32) {
    v16h aq = load_a_frag(qin, D_, row0, kk);
    v16h ax = load_a_frag(xn, D_, row0, kk);
    cq = wmma_f16(aq, load_bt_frag(WtQ, D_, h * 16, kk), cq);
    ck = wmma_f16(aq, load_bt_frag(WtK, D_, h * 16, kk), ck);
    cv = wmma_f16(ax, load_bt_frag(WtV, D_, h * 16, kk), cv);
    cg = wmma_f16(ax, load_bt_frag(WtG, D_, h * 16, kk), cg);
  }
  int n = lane & 15;                 // hd
  int roff = (lane & 16) ? 8 : 0;
  int grp = row0 >> 9;
  int sl0 = row0 & (S_ - 1);
  size_t headbase = ((size_t)grp * H_ + h) * S_;
#pragma unroll
  for (int r = 0; r < 8; ++r) {
    int s = sl0 + r + roff;
    size_t grow = (size_t)row0 + r + roff;
    qb[(headbase + s) * HD_ + n] = (f16)(cq[r] * 0.25f);  // * HD^-0.5
    kb[(headbase + s) * HD_ + n] = (f16)ck[r];
    vT[(((size_t)grp * H_ + h) * HD_ + n) * S_ + s] = (f16)cv[r];
    gb[grow * D_ + h * 16 + n] = (f16)cg[r];
  }
}

// Retention attention: fused decay mask + softmax + mask-renorm + attn@v
// + per-head RMSNorm + SiLU gate.  grid (S/128, H, G), block 256 (8 waves),
// each wave owns one 16-row tile of one (g,h).
// Pass 1 keeps softmax stats LANE-LOCAL (no per-tile cross-lane reductions);
// a single 4-step logsumexp merge combines the 16 column-lanes at the end.
__global__ void k_attn(const f16* __restrict__ qb, const f16* __restrict__ kb,
                       const f16* __restrict__ vT, const f16* __restrict__ gb,
                       const int* __restrict__ coords, f16* __restrict__ og) {
  int g = blockIdx.z, h = blockIdx.y;
  int wave = threadIdx.x >> 5, lane = threadIdx.x & 31;
  int s0 = blockIdx.x * 128 + wave * 16;
  const f16* q = qb + ((size_t)g * H_ + h) * S_ * HD_;
  const f16* k = kb + ((size_t)g * H_ + h) * S_ * HD_;
  const f16* v = vT + ((size_t)g * H_ + h) * HD_ * S_;
  const int* cg = coords + (size_t)g * S_ * 4;
  // log2 of per-head decay factor: mask m = 2^(dist * decay2)
  float decay2 = log2f(1.0f - fast_exp2(-5.0f - (float)h));

  __shared__ float ccx[S_], ccy[S_], ccz[S_];
  __shared__ __align__(16) _Float16 sc[8][16][32];

  // whole block shares the same group: stage its 512 coords in LDS once
  for (int i = threadIdx.x; i < S_; i += 256) {
    ccx[i] = (float)cg[i * 4 + 1];
    ccy[i] = (float)cg[i * 4 + 2];
    ccz[i] = (float)cg[i * 4 + 3];
  }
  __syncthreads();

  int roff = (lane & 16) ? 8 : 0;
  int ncol = lane & 15;
  // row coords for this lane's 8 rows, hoisted to registers
  float rx[8], ry[8], rz[8];
#pragma unroll
  for (int r = 0; r < 8; ++r) {
    int srow = s0 + r + roff;
    rx[r] = ccx[srow]; ry[r] = ccy[srow]; rz[r] = ccz[srow];
  }

  v16h aq = load_a_frag_k16(q, HD_, s0);

  // lane-local online stats (base-2 domain)
  float lmax[8], lsE[8], lsEM[8], lsM[8];
#pragma unroll
  for (int r = 0; r < 8; ++r) {
    lmax[r] = -1e30f; lsE[r] = 0.f; lsEM[r] = 0.f; lsM[r] = 0.f;
  }

  // ---- pass 1: row statistics ----
  for (int t0 = 0; t0 < S_; t0 += 16) {
    v8f c = wmma_f16(aq, load_b_frag_k16(k, t0), zero_v8f());
    int t = t0 + ncol;
    float cx = ccx[t], cy = ccy[t], cz = ccz[t];
#pragma unroll
    for (int r = 0; r < 8; ++r) {
      float x2 = c[r] * LOG2E_;
      float dist = fabsf(rx[r] - cx) + fabsf(ry[r] - cy) + fabsf(rz[r] - cz);
      float m = fast_exp2(dist * decay2);
      float nm = fmaxf(lmax[r], x2);
      float scl = fast_exp2(lmax[r] - nm);
      float e = fast_exp2(x2 - nm);
      lsE[r] = lsE[r] * scl + e;
      lsEM[r] = lsEM[r] * scl + e * m;
      lsM[r] += m;
      lmax[r] = nm;
    }
  }

  // ---- merge the 16 column-lanes (logsumexp combine), once ----
#pragma unroll
  for (int r = 0; r < 8; ++r) {
#pragma unroll
    for (int d = 1; d < 16; d <<= 1) {
      float om = __shfl_xor(lmax[r], d, 16);
      float oE = __shfl_xor(lsE[r], d, 16);
      float oEM = __shfl_xor(lsEM[r], d, 16);
      float oM = __shfl_xor(lsM[r], d, 16);
      float nm = fmaxf(lmax[r], om);
      float s1 = fast_exp2(lmax[r] - nm);
      float s2 = fast_exp2(om - nm);
      lsE[r] = lsE[r] * s1 + oE * s2;
      lsEM[r] = lsEM[r] * s1 + oEM * s2;
      lsM[r] += oM;
      lmax[r] = nm;
    }
  }
  float coef[8];
#pragma unroll
  for (int r = 0; r < 8; ++r) {
    float inv = 1.0f / (lsE[r] * sqrtf(lsM[r]));  // 1/(Z*sqrt(msum))
    float asum = lsEM[r] * inv;                   // row sum of attn
    float denom = fmaxf(asum, 1.0f);              // clip(.,1,inf)
    coef[r] = inv / denom;
  }

  // ---- pass 2: recompute scores, weight, and attn @ v via WMMA ----
  v8f o = zero_v8f();
  for (int t0 = 0; t0 < S_; t0 += 32) {
#pragma unroll
    for (int half = 0; half < 2; ++half) {
      int tt = t0 + half * 16;
      v8f c = wmma_f16(aq, load_b_frag_k16(k, tt), zero_v8f());
      int t = tt + ncol;
      float cx = ccx[t], cy = ccy[t], cz = ccz[t];
#pragma unroll
      for (int r = 0; r < 8; ++r) {
        float dist = fabsf(rx[r] - cx) + fabsf(ry[r] - cy) + fabsf(rz[r] - cz);
        float wgt = fast_exp2(c[r] * LOG2E_ - lmax[r] + dist * decay2) * coef[r];
        sc[wave][r + roff][half * 16 + ncol] = (f16)wgt;
      }
    }
    __syncthreads();   // uniform trip counts; orders LDS write -> read
    v16h aa = load_a_frag(&sc[wave][0][0], 32, 0, 0);
    v16h bv = load_bt_frag(v, S_, 0, t0);   // v^T rows = hd, contiguous in t
    o = wmma_f16(aa, bv, o);
    __syncthreads();
  }

  // ---- epilogue: per-head RMSNorm (over HD) then SiLU(g) gate ----
#pragma unroll
  for (int r = 0; r < 8; ++r) {
    float x = o[r];
    float ss = redsum16(x * x);
    float rn = rsqrtf(ss * (1.0f / HD_) + 1e-6f);
    size_t grow = (size_t)g * S_ + s0 + r + roff;
    float gv = (float)gb[grow * D_ + h * 16 + ncol];
    float sil = gv / (1.0f + fast_exp2(-gv * LOG2E_));
    og[grow * D_ + h * 16 + ncol] = (f16)(x * rn * sil);
  }
}

// x1 = x + og@Wo ; h = RMSNorm(x1, ln2_w). block = one 16-row tile, 8 waves.
__global__ void k_oproj(const f16* __restrict__ og, const f16* __restrict__ WtO,
                        const float* __restrict__ x,
                        const float* __restrict__ ln2w, float* __restrict__ x1,
                        f16* __restrict__ hb) {
  int mt = blockIdx.x;
  int w = threadIdx.x >> 5;
  int lane = threadIdx.x & 31;
  int row0 = mt * 16;
  v8f c = zero_v8f();
#pragma unroll
  for (int kk = 0; kk < D_; kk += 32) {
    v16h a = load_a_frag(og, D_, row0, kk);
    c = wmma_f16(a, load_bt_frag(WtO, D_, w * 16, kk), c);
  }
  __shared__ float xt[16][D_];
  __shared__ float ssp[16][16];
  __shared__ float rs[16];
  int n = w * 16 + (lane & 15);
  int roff = (lane & 16) ? 8 : 0;
#pragma unroll
  for (int r = 0; r < 8; ++r) {
    int rr = r + roff;
    float val = c[r] + x[((size_t)row0 + rr) * D_ + n];
    x1[((size_t)row0 + rr) * D_ + n] = val;
    xt[rr][n] = val;
  }
  __syncthreads();
  int tid = threadIdx.x;
  int row = tid & 15, seg = tid >> 4;
  float s = 0.f;
#pragma unroll
  for (int j = 0; j < 8; ++j) {
    float t = xt[row][seg * 8 + j];
    s += t * t;
  }
  ssp[row][seg] = s;
  __syncthreads();
  if (tid < 16) {
    float t = 0.f;
#pragma unroll
    for (int j = 0; j < 16; ++j) t += ssp[tid][j];
    rs[tid] = rsqrtf(t * (1.0f / D_) + 1e-6f);
  }
  __syncthreads();
#pragma unroll
  for (int j = 0; j < 8; ++j) {
    int col = seg * 8 + j;
    hb[((size_t)row0 + row) * D_ + col] = (f16)(xt[row][col] * rs[row] * ln2w[col]);
  }
}

__device__ __forceinline__ float gelu_tanh(float x) {
  return 0.5f * x * (1.0f + tanhf(0.7978845608028654f * (x + 0.044715f * x * x * x)));
}

// a = gelu(h@W_fc1) * (h@W_gate).  8 waves x 2 N-tiles each (N=256).
__global__ void k_ffn1(const f16* __restrict__ hb, const f16* __restrict__ WtF1,
                       const f16* __restrict__ WtGt, f16* __restrict__ ab) {
  int mt = blockIdx.x;
  int w = threadIdx.x >> 5;
  int lane = threadIdx.x & 31;
  int row0 = mt * 16;
  int n0a = w * 16, n0b = (w + 8) * 16;
  v8f c1a = zero_v8f(), cga = zero_v8f(), c1b = zero_v8f(), cgb = zero_v8f();
#pragma unroll
  for (int kk = 0; kk < D_; kk += 32) {
    v16h a = load_a_frag(hb, D_, row0, kk);
    c1a = wmma_f16(a, load_bt_frag(WtF1, D_, n0a, kk), c1a);
    cga = wmma_f16(a, load_bt_frag(WtGt, D_, n0a, kk), cga);
    c1b = wmma_f16(a, load_bt_frag(WtF1, D_, n0b, kk), c1b);
    cgb = wmma_f16(a, load_bt_frag(WtGt, D_, n0b, kk), cgb);
  }
  int roff = (lane & 16) ? 8 : 0;
#pragma unroll
  for (int r = 0; r < 8; ++r) {
    size_t grow = (size_t)row0 + r + roff;
    ab[grow * FFN_ + n0a + (lane & 15)] = (f16)(gelu_tanh(c1a[r]) * cga[r]);
    ab[grow * FFN_ + n0b + (lane & 15)] = (f16)(gelu_tanh(c1b[r]) * cgb[r]);
  }
}

// out = x1 + a @ W_fc2
__global__ void k_ffn2(const f16* __restrict__ ab, const f16* __restrict__ WtF2,
                       const float* __restrict__ x1, float* __restrict__ out) {
  int mt = blockIdx.x;
  int w = threadIdx.x >> 5;
  int lane = threadIdx.x & 31;
  int row0 = mt * 16;
  v8f c = zero_v8f();
#pragma unroll
  for (int kk = 0; kk < FFN_; kk += 32) {
    v16h a = load_a_frag(ab, FFN_, row0, kk);
    c = wmma_f16(a, load_bt_frag(WtF2, FFN_, w * 16, kk), c);
  }
  int n = w * 16 + (lane & 15);
  int roff = (lane & 16) ? 8 : 0;
#pragma unroll
  for (int r = 0; r < 8; ++r) {
    size_t idx = ((size_t)row0 + r + roff) * D_ + n;
    out[idx] = x1[idx] + c[r];
  }
}

// ---------------------------------------------------------------- launch

extern "C" void kernel_launch(void* const* d_in, const int* in_sizes, int n_in,
                              void* d_out, int out_size, void* d_ws,
                              size_t ws_size, hipStream_t stream) {
  const float* x      = (const float*)d_in[0];
  const float* pe     = (const float*)d_in[1];
  const int*   coords = (const int*)d_in[2];
  const float* Wq     = (const float*)d_in[3];
  const float* Wk     = (const float*)d_in[4];
  const float* Wv     = (const float*)d_in[5];
  const float* Wg     = (const float*)d_in[6];
  const float* Wo     = (const float*)d_in[7];
  const float* ln1w   = (const float*)d_in[8];
  const float* ln2w   = (const float*)d_in[9];
  const float* Wfc1   = (const float*)d_in[10];
  const float* Wgate  = (const float*)d_in[11];
  const float* Wfc2   = (const float*)d_in[12];
  float* out = (float*)d_out;

  char* base = (char*)d_ws;
  size_t off = 0;
  auto alloc = [&](size_t bytes) -> void* {
    void* p = base + off;
    off = (off + bytes + 255) & ~(size_t)255;
    return p;
  };

  f16* WtQ = (f16*)alloc((size_t)D_ * D_ * 2);
  f16* WtK = (f16*)alloc((size_t)D_ * D_ * 2);
  f16* WtV = (f16*)alloc((size_t)D_ * D_ * 2);
  f16* WtG = (f16*)alloc((size_t)D_ * D_ * 2);
  f16* WtO = (f16*)alloc((size_t)D_ * D_ * 2);
  f16* WtF1 = (f16*)alloc((size_t)D_ * FFN_ * 2);
  f16* WtGt = (f16*)alloc((size_t)D_ * FFN_ * 2);
  f16* WtF2 = (f16*)alloc((size_t)FFN_ * D_ * 2);
  f16* xn   = (f16*)alloc((size_t)BM_ * D_ * 2);
  f16* qin  = (f16*)alloc((size_t)BM_ * D_ * 2);
  f16* qb   = (f16*)alloc((size_t)BM_ * D_ * 2);
  f16* kb   = (f16*)alloc((size_t)BM_ * D_ * 2);
  f16* vT   = (f16*)alloc((size_t)BM_ * D_ * 2);
  f16* gb   = (f16*)alloc((size_t)BM_ * D_ * 2);
  f16* og   = (f16*)alloc((size_t)BM_ * D_ * 2);
  f16* hb   = (f16*)alloc((size_t)BM_ * D_ * 2);
  f16* ab   = (f16*)alloc((size_t)BM_ * FFN_ * 2);
  float* x1 = (float*)alloc((size_t)BM_ * D_ * 4);

  // weight transpose+convert
  k_wt<<<(D_ * D_ + 255) / 256, 256, 0, stream>>>(Wq, WtQ, D_, D_);
  k_wt<<<(D_ * D_ + 255) / 256, 256, 0, stream>>>(Wk, WtK, D_, D_);
  k_wt<<<(D_ * D_ + 255) / 256, 256, 0, stream>>>(Wv, WtV, D_, D_);
  k_wt<<<(D_ * D_ + 255) / 256, 256, 0, stream>>>(Wg, WtG, D_, D_);
  k_wt<<<(D_ * D_ + 255) / 256, 256, 0, stream>>>(Wo, WtO, D_, D_);
  k_wt<<<(D_ * FFN_ + 255) / 256, 256, 0, stream>>>(Wfc1, WtF1, D_, FFN_);
  k_wt<<<(D_ * FFN_ + 255) / 256, 256, 0, stream>>>(Wgate, WtGt, D_, FFN_);
  k_wt<<<(FFN_ * D_ + 255) / 256, 256, 0, stream>>>(Wfc2, WtF2, FFN_, D_);

  k_prenorm<<<BM_ / 8, 256, 0, stream>>>(x, pe, ln1w, xn, qin);
  k_qkvg<<<BM_ / 16, 256, 0, stream>>>(qin, xn, WtQ, WtK, WtV, WtG, qb, kb, vT, gb);
  k_attn<<<dim3(S_ / 128, H_, G_), 256, 0, stream>>>(qb, kb, vT, gb, coords, og);
  k_oproj<<<BM_ / 16, 256, 0, stream>>>(og, WtO, x, ln2w, x1, hb);
  k_ffn1<<<BM_ / 16, 256, 0, stream>>>(hb, WtF1, WtGt, ab);
  k_ffn2<<<BM_ / 16, 256, 0, stream>>>(ab, WtF2, x1, out);
}
